// TinyTRM_Inner_17832704213521
// MI455X (gfx1250) — matrix operand order repaired and checked by reading, not compiled, not run on hardware
//
#include <hip/hip_runtime.h>
#include <hip/hip_bf16.h>

#define BATCH 8
#define LSEQ 4096
#define DM 1024
#define HID 512
#define NS 64
#define DS 256
#define TM 32   // rows per macro tile

typedef __attribute__((ext_vector_type(16))) __bf16 v16bf;
typedef __attribute__((ext_vector_type(8)))  __bf16 v8bf;
typedef __attribute__((ext_vector_type(8)))  float  v8f;
typedef __attribute__((ext_vector_type(4)))  unsigned int u32x4;
typedef __attribute__((ext_vector_type(8)))  unsigned int u32x8;

// ---------------------------------------------------------------------------
// WMMA fragment helpers (wave32, v_wmma_f32_16x16x32_bf16)
// A: 16(M) x 32(K). lane l: row = l&15; e<8 -> K = (l>>4)*8 + e ; e>=8 -> K = 16 + (l>>4)*8 + (e-8)
// B: 32(K) x 16(N). lane l: col = l&15; e<8 -> K = (l>>4)*8 + e ; e>=8 -> K = 16 + (l>>4)*8 + (e-8)
// C/D: lane l holds col = l&15, rows = (l>>4)*8 + j, j=0..7
// ---------------------------------------------------------------------------
__device__ inline v16bf load_a_frag(const __bf16* base, int ld) {
  int lane = threadIdx.x & 31;
  const __bf16* row = base + (size_t)(lane & 15) * ld + ((lane >> 4) << 3);
  v8bf lo = *(const v8bf*)(row);
  v8bf hi = *(const v8bf*)(row + 16);
  v16bf r;
#pragma unroll
  for (int e = 0; e < 8; ++e) { r[e] = lo[e]; r[e + 8] = hi[e]; }
  return r;
}

// B stored row-major [N][K]  (i.e. torch weight W[n][k], computing x @ W^T)
__device__ inline v16bf load_b_frag_nk(const __bf16* base, int ld) {
  int lane = threadIdx.x & 31;
  const __bf16* row = base + (size_t)(lane & 15) * ld + ((lane >> 4) << 3);
  v8bf lo = *(const v8bf*)(row);
  v8bf hi = *(const v8bf*)(row + 16);
  v16bf r;
#pragma unroll
  for (int e = 0; e < 8; ++e) { r[e] = lo[e]; r[e + 8] = hi[e]; }
  return r;
}

__device__ inline v8f wmma_bf16(v16bf a, v16bf b, v8f c) {
  return __builtin_amdgcn_wmma_f32_16x16x32_bf16(false, a, false, b, (short)0, c, false, false);
}

__device__ inline float sigmoidf_(float v) { return 1.f / (1.f + __expf(-v)); }
__device__ inline float siluf_(float v)    { return v / (1.f + __expf(-v)); }

// ---------------------------------------------------------------------------
// Tensor Data Mover: async copy of a 2-D bf16 tile (rows x cols, row-major,
// stride == cols) from global memory into LDS.  D# layout per CDNA5 ISA
// (CDNA5 ISA): group0 = {count/type/addr}, group1 = dims.
// Issued by one wave; completion via TENSORcnt.
// ---------------------------------------------------------------------------
__device__ inline void tdm_load_2d_bf16(unsigned lds_addr, const __bf16* gptr,
                                        unsigned rows, unsigned cols) {
  unsigned long long ga = (unsigned long long)(uintptr_t)gptr;
  u32x4 g0;
  g0[0] = 1u;                                   // count=1, is_restore=0, gather=0
  g0[1] = lds_addr;                             // lds_addr [63:32]
  g0[2] = (unsigned)ga;                         // global_addr [95:64]
  g0[3] = (unsigned)(ga >> 32) | 0x80000000u;   // global_addr hi + type=2 [127:126]
  u32x8 g1;
  g1[0] = 0x00010000u;                          // wg_mask=0, data_size=1 (2 bytes)
  g1[1] = (cols & 0xffffu) << 16;               // tensor_dim0 low16  (bits 63:48)
  g1[2] = (rows & 0xffffu) << 16;               // dim0 hi=0 | tensor_dim1 low16
  g1[3] = (cols & 0xffffu) << 16;               // dim1 hi=0 | tile_dim0
  g1[4] = (rows & 0xffffu);                     // tile_dim1 | tile_dim2=0
  g1[5] = cols;                                 // tensor_dim0_stride [191:160]
  g1[6] = 0u;                                   // stride hi | dim1_stride lo
  g1[7] = 0u;
  u32x4 g2 = {0u, 0u, 0u, 0u};                  // tensor_dim2.. unused (2-D tile)
  u32x4 g3 = {0u, 0u, 0u, 0u};
  asm volatile("tensor_load_to_lds %0, %1, %2, %3"
               :: "s"(g0), "s"(g1), "s"(g2), "s"(g3)
               : "memory");
}

// ---------------------------------------------------------------------------
// K0: fp32 -> bf16 weight conversion
// ---------------------------------------------------------------------------
__global__ __launch_bounds__(256) void cvt_kernel(const float* __restrict__ src,
                                                  __bf16* __restrict__ dst, int n) {
  int i = blockIdx.x * 256 + threadIdx.x;
  if (i < n) dst[i] = (__bf16)src[i];
}

// ---------------------------------------------------------------------------
// K1: slot self-attention, one block per (batch, head); accumulates out-proj
// ---------------------------------------------------------------------------
__global__ __launch_bounds__(256) void mha_kernel(const float* __restrict__ slots,
                                                  const float* __restrict__ Wqkv,
                                                  const float* __restrict__ bqkv,
                                                  const float* __restrict__ Wo,
                                                  float* __restrict__ accum) {
  __shared__ float qh[NS * 64], kh[NS * 64], vh[NS * 64], sc[NS * 64];
  int b = blockIdx.x >> 2, h = blockIdx.x & 3, t = threadIdx.x;
  const float* sl = slots + (size_t)b * NS * DS;
  for (int idx = t; idx < NS * 64; idx += 256) {
    int s = idx >> 6, d = idx & 63;
    const float* srow = sl + s * DS;
    const float* wq = Wqkv + (size_t)(h * 64 + d) * DS;
    const float* wk = Wqkv + (size_t)(DS + h * 64 + d) * DS;
    const float* wv = Wqkv + (size_t)(2 * DS + h * 64 + d) * DS;
    float aq = bqkv[h * 64 + d], ak = bqkv[DS + h * 64 + d], av = bqkv[2 * DS + h * 64 + d];
    for (int c = 0; c < DS; ++c) { float sv = srow[c]; aq += sv * wq[c]; ak += sv * wk[c]; av += sv * wv[c]; }
    qh[idx] = aq; kh[idx] = ak; vh[idx] = av;
  }
  __syncthreads();
  for (int idx = t; idx < NS * NS; idx += 256) {
    int i = idx >> 6, j = idx & 63;
    float a = 0.f;
    for (int d = 0; d < 64; ++d) a += qh[i * 64 + d] * kh[j * 64 + d];
    sc[idx] = a * 0.125f;
  }
  __syncthreads();
  if (t < NS) {
    float mx = -1e30f;
    for (int j = 0; j < 64; ++j) mx = fmaxf(mx, sc[t * 64 + j]);
    float sum = 0.f;
    for (int j = 0; j < 64; ++j) { float e = __expf(sc[t * 64 + j] - mx); sc[t * 64 + j] = e; sum += e; }
    float inv = 1.f / sum;
    for (int j = 0; j < 64; ++j) sc[t * 64 + j] *= inv;
  }
  __syncthreads();
  for (int idx = t; idx < NS * 64; idx += 256) {       // attn @ v -> reuse qh
    int s = idx >> 6, d = idx & 63;
    float a = 0.f;
    for (int j = 0; j < 64; ++j) a += sc[s * 64 + j] * vh[j * 64 + d];
    qh[idx] = a;
  }
  __syncthreads();
  for (int idx = t; idx < NS * DS; idx += 256) {       // out-proj slice for this head
    int s = idx >> 8, dc = idx & 255;
    const float* wo = Wo + (size_t)dc * DS + h * 64;
    float a = 0.f;
    for (int d = 0; d < 64; ++d) a += qh[s * 64 + d] * wo[d];
    atomicAdd(&accum[((size_t)b * NS + s) * DS + dc], a);
  }
}

// K1b: residual + LayerNorm -> slots2.  grid = B*NS rows, 256 threads/row
__global__ __launch_bounds__(256) void ln_kernel(const float* __restrict__ slots,
                                                 const float* __restrict__ accum,
                                                 const float* __restrict__ bo,
                                                 const float* __restrict__ ln_g,
                                                 const float* __restrict__ ln_b,
                                                 float* __restrict__ slots2) {
  __shared__ float red[256];
  __shared__ float s_mean, s_rstd;
  int row = blockIdx.x, t = threadIdx.x;
  float y = slots[(size_t)row * DS + t] + accum[(size_t)row * DS + t] + bo[t];
  red[t] = y; __syncthreads();
  for (int s = 128; s > 0; s >>= 1) { if (t < s) red[t] += red[t + s]; __syncthreads(); }
  if (t == 0) s_mean = red[0] * (1.f / DS);
  __syncthreads();
  float d = y - s_mean;
  red[t] = d * d; __syncthreads();
  for (int s = 128; s > 0; s >>= 1) { if (t < s) red[t] += red[t + s]; __syncthreads(); }
  if (t == 0) s_rstd = rsqrtf(red[0] * (1.f / DS) + 1e-5f);
  __syncthreads();
  slots2[(size_t)row * DS + t] = d * s_rstd * ln_g[t] + ln_b[t];
}

// K2: k = slots2@rk^T+b (bf16), vT = (slots2@rv^T+b)^T (bf16, [DM][NS] per batch)
__global__ __launch_bounds__(256) void kv_kernel(const float* __restrict__ slots2,
                                                 const float* __restrict__ rk_W, const float* __restrict__ rk_b,
                                                 const float* __restrict__ rv_W, const float* __restrict__ rv_b,
                                                 __bf16* __restrict__ kbf, __bf16* __restrict__ vtbf) {
  int b = blockIdx.x, t = threadIdx.x;
  const float* sl = slots2 + (size_t)b * NS * DS;
  for (int idx = t; idx < NS * DS; idx += 256) {
    int s = idx >> 8, d = idx & 255;
    const float* w = rk_W + (size_t)d * DS; const float* sr = sl + s * DS;
    float a = rk_b[d];
    for (int c = 0; c < DS; ++c) a += sr[c] * w[c];
    kbf[((size_t)b * NS + s) * DS + d] = (__bf16)a;
  }
  for (int idx = t; idx < NS * DM; idx += 256) {
    int s = idx >> 10, d = idx & 1023;
    const float* w = rv_W + (size_t)d * DS; const float* sr = sl + s * DS;
    float a = rv_b[d];
    for (int c = 0; c < DS; ++c) a += sr[c] * w[c];
    vtbf[((size_t)b * DM + d) * NS + s] = (__bf16)a;   // transposed store
  }
}

// ---------------------------------------------------------------------------
// K3: read path (WMMA). grid = B*L/TM blocks, 256 threads (8 waves).
// k-matrix tile is DMA'd into LDS by the Tensor Data Mover, overlapped with
// the x-tile staging and the rg/route/q GEMMs; waited via TENSORcnt.
// ---------------------------------------------------------------------------
__global__ __launch_bounds__(256) void read_kernel(const float* __restrict__ x,
                                                   const __bf16* __restrict__ w_rg1,
                                                   const float* __restrict__ rg_b1,
                                                   const float* __restrict__ rg_w2,
                                                   const float* __restrict__ rg_b2,
                                                   const __bf16* __restrict__ w_rr,
                                                   const float* __restrict__ rr_b,
                                                   const __bf16* __restrict__ w_rq,
                                                   const float* __restrict__ rq_b,
                                                   const __bf16* __restrict__ kbf,
                                                   const float* __restrict__ gamma,
                                                   float* __restrict__ readgate,
                                                   __bf16* __restrict__ rwbf) {
  __shared__ __bf16 xs[TM * DM];    // 64 KB
  __shared__ __bf16 ks[NS * DS];    // 32 KB (k matrix, TDM destination)
  __shared__ __bf16 qs[TM * DS];    // 16 KB
  __shared__ float logit[TM * NS];  // 8 KB
  __shared__ float accg[TM];

  int tile = blockIdx.x;
  int b = tile / (LSEQ / TM);
  int r0 = (tile % (LSEQ / TM)) * TM;
  const float* xrow = x + ((size_t)b * LSEQ + r0) * DM;
  int t = threadIdx.x;

  // kick off async TDM copy of k[b] (64 x 256 bf16) into LDS from wave 0
  if ((t >> 5) == 0) {
    tdm_load_2d_bf16((unsigned)(size_t)ks, kbf + (size_t)b * NS * DS, NS, DS);
  }

  for (int i = t; i < TM * DM; i += 256) xs[i] = (__bf16)xrow[i];
  if (t < TM) accg[t] = 0.f;
  __syncthreads();

  int w = t >> 5, lane = t & 31;
  int mi = w >> 2, ni = w & 3;
  int crow0 = mi * 16 + (lane >> 4) * 8;
  int ccol = lane & 15;

  // ---- read gate: silu(x@W1^T+b1) . w2 ----
  for (int nc = 0; nc < 8; ++nc) {
    v8f c = {};
    int nbase = nc * 64 + ni * 16;
    for (int kb = 0; kb < DM; kb += 32) {
      v16bf a = load_a_frag(xs + mi * 16 * DM + kb, DM);
      v16bf bb = load_b_frag_nk(w_rg1 + (size_t)nbase * DM + kb, DM);
      c = wmma_bf16(a, bb, c);
    }
#pragma unroll
    for (int j = 0; j < 8; ++j) {
      int col = nbase + ccol;
      float h = siluf_(c[j] + rg_b1[col]);
      atomicAdd(&accg[crow0 + j], h * rg_w2[col]);
    }
  }

  // ---- route logits (N=64) ----
  {
    v8f c = {};
    int nbase = ni * 16;
    for (int kb = 0; kb < DM; kb += 32) {
      v16bf a = load_a_frag(xs + mi * 16 * DM + kb, DM);
      v16bf bb = load_b_frag_nk(w_rr + (size_t)nbase * DM + kb, DM);
      c = wmma_bf16(a, bb, c);
    }
    float g = gamma[0];
#pragma unroll
    for (int j = 0; j < 8; ++j) {
      int col = nbase + ccol;
      logit[(crow0 + j) * NS + col] = 0.7f * g * (c[j] + rr_b[col]);
    }
  }

  // ---- q (N=256) ----
  for (int nc = 0; nc < 4; ++nc) {
    v8f c = {};
    int nbase = nc * 64 + ni * 16;
    for (int kb = 0; kb < DM; kb += 32) {
      v16bf a = load_a_frag(xs + mi * 16 * DM + kb, DM);
      v16bf bb = load_b_frag_nk(w_rq + (size_t)nbase * DM + kb, DM);
      c = wmma_bf16(a, bb, c);
    }
#pragma unroll
    for (int j = 0; j < 8; ++j) {
      int col = nbase + ccol;
      qs[(crow0 + j) * DS + col] = (__bf16)(c[j] + rq_b[col]);
    }
  }
  // wait for TDM copy of k (wave 0 owns the TENSORcnt), then barrier
  if ((t >> 5) == 0) __builtin_amdgcn_s_wait_tensorcnt(0);
  __syncthreads();

  // ---- content logits: q @ k^T / 16, blended 0.3 (k read from LDS) ----
  {
    v8f c = {};
    int nbase = ni * 16;
    for (int kb = 0; kb < DS; kb += 32) {
      v16bf a = load_a_frag(qs + mi * 16 * DS + kb, DS);
      v16bf bb = load_b_frag_nk(ks + (size_t)nbase * DS + kb, DS);
      c = wmma_bf16(a, bb, c);
    }
#pragma unroll
    for (int j = 0; j < 8; ++j) {
      int col = nbase + ccol;
      logit[(crow0 + j) * NS + col] += c[j] * (0.3f / 16.f);
    }
  }
  __syncthreads();

  // ---- softmax rows + gate finalize ----
  if (t < TM) {
    readgate[(size_t)b * LSEQ + r0 + t] = sigmoidf_(accg[t] + rg_b2[0]);
    float mx = -1e30f;
    for (int s = 0; s < NS; ++s) mx = fmaxf(mx, logit[t * NS + s]);
    float sum = 0.f;
    for (int s = 0; s < NS; ++s) { float e = __expf(logit[t * NS + s] - mx); logit[t * NS + s] = e; sum += e; }
    float inv = 1.f / sum;
    __bf16* dst = rwbf + ((size_t)b * LSEQ + r0 + t) * NS;
    for (int s = 0; s < NS; ++s) dst[s] = (__bf16)(logit[t * NS + s] * inv);
  }
}

// ---------------------------------------------------------------------------
// K4: context + fuse gate (WMMA). grid = B*L/TM blocks, 256 threads.
// v is consumed transposed ([DM][NS]) so all B-fragments are contiguous.
// ---------------------------------------------------------------------------
__global__ __launch_bounds__(256) void fuse_kernel(const float* __restrict__ x,
                                                   const __bf16* __restrict__ vtbf,
                                                   const __bf16* __restrict__ w_fg,   // [1024][2048] bf16
                                                   const float* __restrict__ fg_b,
                                                   const float* __restrict__ readgate,
                                                   const __bf16* __restrict__ rwbf,
                                                   float* __restrict__ xfused,
                                                   float* __restrict__ xpool) {
  __shared__ __bf16 xs[TM * DM];    // 64 KB
  __shared__ __bf16 cs[TM * DM];    // 64 KB
  __shared__ __bf16 rws[TM * NS];   // 4 KB
  __shared__ float rg[TM];
  __shared__ float pool[DM];        // 4 KB

  int tile = blockIdx.x;
  int b = tile / (LSEQ / TM);
  int r0 = (tile % (LSEQ / TM)) * TM;
  const float* xrow = x + ((size_t)b * LSEQ + r0) * DM;
  int t = threadIdx.x;

  for (int i = t; i < TM * DM; i += 256) xs[i] = (__bf16)xrow[i];
  for (int i = t; i < TM * NS; i += 256) rws[i] = rwbf[((size_t)b * LSEQ + r0) * NS + i];
  if (t < TM) rg[t] = readgate[(size_t)b * LSEQ + r0 + t];
  for (int i = t; i < DM; i += 256) pool[i] = 0.f;
  __syncthreads();

  int w = t >> 5, lane = t & 31;
  int mi = w >> 2, ni = w & 3;
  int crow0 = mi * 16 + (lane >> 4) * 8;
  int ccol = lane & 15;

  // ---- context = (rw @ v) * read_gate ;  B = vT in NK layout ----
  const __bf16* vb = vtbf + (size_t)b * DM * NS;
  for (int nc = 0; nc < 16; ++nc) {
    v8f c = {};
    int nbase = nc * 64 + ni * 16;
    for (int kb = 0; kb < NS; kb += 32) {
      v16bf a = load_a_frag(rws + mi * 16 * NS + kb, NS);
      v16bf bb = load_b_frag_nk(vb + (size_t)nbase * NS + kb, NS);
      c = wmma_bf16(a, bb, c);
    }
#pragma unroll
    for (int j = 0; j < 8; ++j) {
      int row = crow0 + j, col = nbase + ccol;
      cs[row * DM + col] = (__bf16)(c[j] * rg[row]);
    }
  }
  __syncthreads();

  // ---- gate = sigmoid([x, ctx] @ fg_W^T + b); x_fused = g*ctx + (1-g)*x ----
  float* xfrow = xfused + ((size_t)b * LSEQ + r0) * DM;
  for (int nc = 0; nc < 16; ++nc) {
    v8f c = {};
    int nbase = nc * 64 + ni * 16;
    for (int kb = 0; kb < DM; kb += 32) {
      v16bf a = load_a_frag(xs + mi * 16 * DM + kb, DM);
      v16bf bb = load_b_frag_nk(w_fg + (size_t)nbase * (2 * DM) + kb, 2 * DM);
      c = wmma_bf16(a, bb, c);
    }
    for (int kb = 0; kb < DM; kb += 32) {
      v16bf a = load_a_frag(cs + mi * 16 * DM + kb, DM);
      v16bf bb = load_b_frag_nk(w_fg + (size_t)nbase * (2 * DM) + DM + kb, 2 * DM);
      c = wmma_bf16(a, bb, c);
    }
#pragma unroll
    for (int j = 0; j < 8; ++j) {
      int row = crow0 + j, col = nbase + ccol;
      float g = sigmoidf_(c[j] + fg_b[col]);
      float ctx = (float)cs[row * DM + col];
      float xv = xrow[(size_t)row * DM + col];
      float xf = g * ctx + (1.f - g) * xv;
      xfrow[(size_t)row * DM + col] = xf;
      atomicAdd(&pool[col], xf);
    }
  }
  __syncthreads();
  for (int i = t; i < DM; i += 256) atomicAdd(&xpool[(size_t)b * DM + i], pool[i]);
}

// ---------------------------------------------------------------------------
// K5: write path (tiny). grid = B blocks.
// ---------------------------------------------------------------------------
__global__ __launch_bounds__(256) void write_kernel(const float* __restrict__ xpool,
                                                    const float* __restrict__ usage,
                                                    const float* __restrict__ slots2,
                                                    const float* __restrict__ wg_W1, const float* __restrict__ wg_b1,
                                                    const float* __restrict__ wg_W2, const float* __restrict__ wg_b2,
                                                    const float* __restrict__ wi_W1, const float* __restrict__ wi_b1,
                                                    const float* __restrict__ wi_W2, const float* __restrict__ wi_b2,
                                                    const float* __restrict__ wr_W, const float* __restrict__ wr_b,
                                                    const float* __restrict__ wc_W, const float* __restrict__ wc_b,
                                                    const float* __restrict__ gamma,
                                                    float* __restrict__ out_slots,
                                                    float* __restrict__ out_usage) {
  __shared__ float xp[DM];
  __shared__ float red[256];
  __shared__ float gates[2];
  __shared__ float wp[NS];
  __shared__ float content[DS];
  int b = blockIdx.x, t = threadIdx.x;
  for (int i = t; i < DM; i += 256) xp[i] = xpool[(size_t)b * DM + i] * (1.f / LSEQ);
  __syncthreads();

  for (int pass = 0; pass < 2; ++pass) {
    const float* W1 = pass ? wi_W1 : wg_W1;
    const float* b1 = pass ? wi_b1 : wg_b1;
    const float* W2 = pass ? wi_W2 : wg_W2;
    const float* b2 = pass ? wi_b2 : wg_b2;
    float part = 0.f;
    for (int hx = t; hx < HID; hx += 256) {
      const float* wr = W1 + (size_t)hx * DM;
      float a = b1[hx];
      for (int c = 0; c < DM; ++c) a += xp[c] * wr[c];
      part += siluf_(a) * W2[hx];
    }
    red[t] = part; __syncthreads();
    for (int s = 128; s > 0; s >>= 1) { if (t < s) red[t] += red[t + s]; __syncthreads(); }
    if (t == 0) gates[pass] = sigmoidf_(red[0] + b2[0]);
    __syncthreads();
  }

  if (t < NS) {
    const float* wr = wr_W + (size_t)t * DM;
    float a = wr_b[t];
    for (int c = 0; c < DM; ++c) a += xp[c] * wr[c];
    wp[t] = a * gamma[0] - 0.5f * usage[(size_t)b * NS + t];
  }
  __syncthreads();
  if (t == 0) {
    float mx = -1e30f;
    for (int s = 0; s < NS; ++s) mx = fmaxf(mx, wp[s]);
    float sum = 0.f;
    for (int s = 0; s < NS; ++s) { float e = __expf(wp[s] - mx); wp[s] = e; sum += e; }
    float inv = 1.f / sum;
    for (int s = 0; s < NS; ++s) wp[s] *= inv;
  }
  __syncthreads();
  if (t < DS) {
    const float* wr = wc_W + (size_t)t * DM;
    float a = wc_b[t];
    for (int c = 0; c < DM; ++c) a += xp[c] * wr[c];
    content[t] = a;
  }
  __syncthreads();
  float score = gates[0] * gates[1];
  for (int i = t; i < NS * DS; i += 256) {
    int s = i >> 8, d = i & 255;
    float uw = score * wp[s];
    out_slots[(size_t)b * NS * DS + i] = (1.f - uw) * slots2[(size_t)b * NS * DS + i] + uw * content[d];
  }
  if (t < NS) out_usage[(size_t)b * NS + t] = 0.99f * usage[(size_t)b * NS + t] + score * wp[t];
}

// ---------------------------------------------------------------------------
extern "C" void kernel_launch(void* const* d_in, const int* in_sizes, int n_in,
                              void* d_out, int out_size, void* d_ws, size_t ws_size,
                              hipStream_t stream) {
  const float* x      = (const float*)d_in[0];
  const float* slots  = (const float*)d_in[1];
  const float* usage  = (const float*)d_in[2];
  const float* rg_W1  = (const float*)d_in[3];
  const float* rg_b1  = (const float*)d_in[4];
  const float* rg_W2  = (const float*)d_in[5];
  const float* rg_b2  = (const float*)d_in[6];
  const float* rr_W   = (const float*)d_in[7];
  const float* rr_b   = (const float*)d_in[8];
  const float* rq_W   = (const float*)d_in[9];
  const float* rq_b   = (const float*)d_in[10];
  const float* rk_W   = (const float*)d_in[11];
  const float* rk_b   = (const float*)d_in[12];
  const float* rv_W   = (const float*)d_in[13];
  const float* rv_b   = (const float*)d_in[14];
  const float* wg_W1  = (const float*)d_in[15];
  const float* wg_b1  = (const float*)d_in[16];
  const float* wg_W2  = (const float*)d_in[17];
  const float* wg_b2  = (const float*)d_in[18];
  const float* wi_W1  = (const float*)d_in[19];
  const float* wi_b1  = (const float*)d_in[20];
  const float* wi_W2  = (const float*)d_in[21];
  const float* wi_b2  = (const float*)d_in[22];
  const float* wr_W   = (const float*)d_in[23];
  const float* wr_b   = (const float*)d_in[24];
  const float* wc_W   = (const float*)d_in[25];
  const float* wc_b   = (const float*)d_in[26];
  const float* gamma  = (const float*)d_in[27];
  const float* fg_W   = (const float*)d_in[28];
  const float* fg_b   = (const float*)d_in[29];
  const float* mhaW   = (const float*)d_in[30];
  const float* mhab   = (const float*)d_in[31];
  const float* mhaWo  = (const float*)d_in[32];
  const float* mhabo  = (const float*)d_in[33];
  const float* ln_g   = (const float*)d_in[34];
  const float* ln_b   = (const float*)d_in[35];

  // ---- workspace layout ----
  char* ws = (char*)d_ws;
  size_t off = 0;
  auto alloc = [&](size_t bytes) { char* p = ws + off; off = (off + bytes + 255) & ~(size_t)255; return p; };
  __bf16* wrg1_bf = (__bf16*)alloc((size_t)HID * DM * 2);
  __bf16* wrr_bf  = (__bf16*)alloc((size_t)NS * DM * 2);
  __bf16* wrq_bf  = (__bf16*)alloc((size_t)DS * DM * 2);
  __bf16* wfg_bf  = (__bf16*)alloc((size_t)DM * 2 * DM * 2);
  float*  accum   = (float*)alloc((size_t)BATCH * NS * DS * 4);
  float*  slots2  = (float*)alloc((size_t)BATCH * NS * DS * 4);
  __bf16* kbf     = (__bf16*)alloc((size_t)BATCH * NS * DS * 2);
  __bf16* vtbf    = (__bf16*)alloc((size_t)BATCH * NS * DM * 2);
  float*  rgate   = (float*)alloc((size_t)BATCH * LSEQ * 4);
  __bf16* rwbf    = (__bf16*)alloc((size_t)BATCH * LSEQ * NS * 2);
  float*  xpool   = (float*)alloc((size_t)BATCH * DM * 4);

  float* out_xf    = (float*)d_out;
  float* out_slots = out_xf + (size_t)BATCH * LSEQ * DM;
  float* out_usage = out_slots + (size_t)BATCH * NS * DS;

  hipMemsetAsync(accum, 0, (size_t)BATCH * NS * DS * 4, stream);
  hipMemsetAsync(xpool, 0, (size_t)BATCH * DM * 4, stream);

  // weight conversion
  cvt_kernel<<<(HID * DM) / 256, 256, 0, stream>>>(rg_W1, wrg1_bf, HID * DM);
  cvt_kernel<<<(NS * DM) / 256, 256, 0, stream>>>(rr_W, wrr_bf, NS * DM);
  cvt_kernel<<<(DS * DM) / 256, 256, 0, stream>>>(rq_W, wrq_bf, DS * DM);
  cvt_kernel<<<(DM * 2 * DM) / 256, 256, 0, stream>>>(fg_W, wfg_bf, DM * 2 * DM);

  // slot attention + LN
  mha_kernel<<<BATCH * 4, 256, 0, stream>>>(slots, mhaW, mhab, mhaWo, accum);
  ln_kernel<<<BATCH * NS, 256, 0, stream>>>(slots, accum, mhabo, ln_g, ln_b, slots2);
  kv_kernel<<<BATCH, 256, 0, stream>>>(slots2, rk_W, rk_b, rv_W, rv_b, kbf, vtbf);

  // read path (WMMA + TDM)
  read_kernel<<<BATCH * (LSEQ / TM), 256, 0, stream>>>(x, wrg1_bf, rg_b1, rg_W2, rg_b2,
                                                       wrr_bf, rr_b, wrq_bf, rq_b,
                                                       kbf, gamma, rgate, rwbf);
  // context + fuse (WMMA)
  fuse_kernel<<<BATCH * (LSEQ / TM), 256, 0, stream>>>(x, vtbf, wfg_bf, fg_b, rgate, rwbf,
                                                       out_xf, xpool);
  // write path
  write_kernel<<<BATCH, 256, 0, stream>>>(xpool, usage, slots2,
                                          wg_W1, wg_b1, wg_W2, wg_b2,
                                          wi_W1, wi_b1, wi_W2, wi_b2,
                                          wr_W, wr_b, wc_W, wc_b,
                                          gamma, out_slots, out_usage);
  (void)in_sizes; (void)n_in; (void)out_size; (void)ws_size;
}
